// Pre_processing_6588479832202
// MI455X (gfx1250) — matrix operand order repaired and checked by reading, not compiled, not run on hardware
//
#include <hip/hip_runtime.h>
#include <hip/hip_bf16.h>

// ---------------------------------------------------------------------------
// CDNA5 (gfx1250) implementation: stereo cost-volume pre-processing.
//   - bf16 WMMA (v_wmma_f32_16x16x32_bf16) for both conv GEMMs
//   - TDM (tensor_load_to_lds) for conv1 weight tiles, with descriptor padding
//     reproducing the bank-conflict-free LDS stride
//   - global_load_async_to_lds_b32/b128 for activation / volume-row staging
//   - f32 VALU for BN / softmax / weighted unfold / correlation
// ---------------------------------------------------------------------------

typedef __attribute__((ext_vector_type(16))) __bf16 v16bf;
typedef __attribute__((ext_vector_type(8)))  float  v8f;
typedef unsigned int uint32x4 __attribute__((ext_vector_type(4)));
typedef int          int32x4  __attribute__((ext_vector_type(4)));
typedef int          int32x8  __attribute__((ext_vector_type(8)));

union Frag16 { v16bf v; unsigned u[8]; };

// Problem constants (from the reference)
#define BB    4
#define ICG   96
#define ICC   12
#define HIMG  160
#define WIMG  288
#define HO    80
#define WO    144
#define MTOT  (BB*HO*WO)      // 46080 = 720*64 = 1440*32
#define DISP  24

__device__ __forceinline__ unsigned short f2bf(float f) {
  union { float f; unsigned u; } v; v.f = f;
  unsigned u = v.u;
  u += 0x7FFFu + ((u >> 16) & 1u);     // round-to-nearest-even
  return (unsigned short)(u >> 16);
}

// Low 32 bits of a generic pointer to __shared__ == wave-relative LDS byte
// address (ISA 10.2: LDS aperture).
__device__ __forceinline__ unsigned lds_addr32(const void* p) {
  return (unsigned)(size_t)p;
}

// Per-lane async global->LDS copies (ASYNCcnt).
__device__ __forceinline__ void async_b32(unsigned lds, const void* g) {
  asm volatile("global_load_async_to_lds_b32 %0, %1, off" :: "v"(lds), "v"(g) : "memory");
}
__device__ __forceinline__ void async_b128(unsigned lds, const void* g) {
  asm volatile("global_load_async_to_lds_b128 %0, %1, off" :: "v"(lds), "v"(g) : "memory");
}
__device__ __forceinline__ void wait_async0() {
  asm volatile("s_wait_asynccnt 0x0" ::: "memory");
}

// TDM: 2D tile load (tile_dim0 x tile_dim1 bf16 elems, row stride = stride0
// elems) into LDS with +1 dword of padding after every 16 dwords (64B) stored,
// i.e. an LDS row stride of 34 halves for 32-element rows. amdgpu-toolchain
// 6-arg builtin; descriptor bitfields per CDNA5 ISA.
__device__ __forceinline__ void tdm_load_tile_pad34(unsigned lds, const void* gsrc,
                                                    unsigned tile_d0, unsigned tile_d1,
                                                    unsigned stride0) {
  unsigned long long ga = (unsigned long long)(size_t)gsrc;
  uint32x4 g0;
  g0[0] = 1u;                                          // count=1, user mode
  g0[1] = lds;                                         // lds_addr (bytes)
  g0[2] = (unsigned)(ga & 0xFFFFFFFFu);                // global_addr[31:0]
  g0[3] = (unsigned)((ga >> 32) & 0x1FFFFFFu) | (2u << 30);  // addr[56:32] | type=2
  int32x8 g1;
  g1[0] = (int)((1u << 16)      // data_size = 2 bytes
              | (1u << 20)      // pad_enable
              | (3u << 22));    // pad_interval: 16 dwords; pad_amount=0 -> 1 dword
  g1[1] = (int)(tile_d0 << 16);               // tensor_dim0[15:0] (== tile_d0)
  g1[2] = (int)(tile_d1 << 16);               // tensor_dim0 hi | tensor_dim1 lo
  g1[3] = (int)(tile_d0 << 16);               // tensor_dim1 hi | tile_dim0
  g1[4] = (int)tile_d1;                       // tile_dim1 | tile_dim2=0
  g1[5] = (int)stride0;                       // tensor_dim0_stride[31:0]
  g1[6] = 0;
  g1[7] = 0;
  int32x4 z4 = {0, 0, 0, 0};
  int32x8 z8 = {0, 0, 0, 0, 0, 0, 0, 0};
  __builtin_amdgcn_tensor_load_to_lds(g0, g1, z4, z4, z8, 0);
}

// ---------------------------------------------------------------------------
// Weight convert + pad to bf16, layout [OCP][KDP] ([oc][k]: N-major,
// K-contiguous -> exactly what the B-fragment loads want).
// ---------------------------------------------------------------------------
__global__ void convert_pad_bf16(const float* __restrict__ src,
                                 unsigned short* __restrict__ dst,
                                 int OC, int KD, int KDP, int total) {
  for (int i = blockIdx.x * blockDim.x + threadIdx.x; i < total;
       i += gridDim.x * blockDim.x) {
    int oc = i / KDP;
    int k  = i - oc * KDP;
    float v = (oc < OC && k < KD) ? src[oc * KD + k] : 0.0f;
    dst[i] = f2bf(v);
  }
}

// ---------------------------------------------------------------------------
// Conv1 (3x3, stride 2, pad 1) as implicit GEMM + BN + leaky ReLU.
//   block: 256 threads = 8 waves, tile 64(M) x 64(N); each wave owns a
//   16x32 strip -> 1 A fragment, 2 B fragments, 2 WMMAs per K-step.
//   A: im2col gathered into LDS (bf16).  B: TDM tile load with pad-34 stride.
// Output: y[m][OCP] bf16 (activation-major == conv2 A layout); padded
// channels written as zero.
// ---------------------------------------------------------------------------
template <int IC, int OC, int OCP, int KD, int KDP>
__global__ __launch_bounds__(256)
void conv1_bn_lrelu_wmma(const float* __restrict__ x,
                         const unsigned short* __restrict__ wt,  // [OCP][KDP]
                         const float* __restrict__ bn_g, const float* __restrict__ bn_b,
                         const float* __restrict__ bn_m, const float* __restrict__ bn_v,
                         unsigned short* __restrict__ y) {
  constexpr int KSTEPS = KDP / 32;
  __shared__ unsigned short As[64 * 34];  // stride 34 halves: conflict-free
  __shared__ unsigned short Bs[64 * 34];  // Bs[n][k], stride 34 (TDM pad)

  const int tid  = threadIdx.x;
  const int lane = tid & 31;
  const int wave = tid >> 5;
  const int wm   = wave & 3;   // M subtile 0..3 (16 rows each)
  const int wn   = wave >> 2;  // N strip 0..1 (32 cols each)
  const int m0   = blockIdx.x * 64;
  const int n0   = blockIdx.y * 64;
  const int half = lane >> 4;
  const int l16  = lane & 15;

  v8f acc0 = {}, acc1 = {};

  for (int ks = 0; ks < KSTEPS; ++ks) {
    // ---- B tile via Tensor Data Mover (one wave issues; TENSORcnt) ----
    if (wave == 0) {
      tdm_load_tile_pad34(lds_addr32(&Bs[0]),
                          wt + (size_t)n0 * KDP + ks * 32,
                          /*tile_d0=*/32, /*tile_d1=*/64, /*stride0=*/KDP);
    }
    // ---- A tile: im2col gather, f32 -> bf16 ----
    for (int i = tid; i < 64 * 32; i += 256) {
      int r = i >> 5, c = i & 31;
      int k = ks * 32 + c;
      float val = 0.0f;
      if (k < KD) {
        int ic = k / 9, kk = k - ic * 9;
        int ky = kk / 3, kx = kk - ky * 3;
        int m   = m0 + r;
        int n   = m / (HO * WO);
        int rem = m - n * (HO * WO);
        int ho  = rem / WO;
        int wo  = rem - ho * WO;
        int ih  = 2 * ho - 1 + ky;
        int iw  = 2 * wo - 1 + kx;
        if ((unsigned)ih < (unsigned)HIMG && (unsigned)iw < (unsigned)WIMG)
          val = x[((n * IC + ic) * HIMG + ih) * WIMG + iw];
      }
      As[r * 34 + c] = f2bf(val);
    }
    __builtin_amdgcn_s_wait_tensorcnt(0);
    __syncthreads();

    // ---- fragment loads (contiguous k = 16*half + s, A and B alike) + 2 WMMAs ----
    Frag16 a, b0, b1;
    const unsigned short* arow  = &As[(wm * 16 + l16) * 34];
    const unsigned short* brow0 = &Bs[(wn * 32 + l16) * 34];
    const unsigned short* brow1 = &Bs[(wn * 32 + 16 + l16) * 34];
#pragma unroll
    for (int i = 0; i < 4; ++i) {
      a.u[i]     = *(const unsigned*)(arow + half * 16 + 2 * i);
      a.u[4 + i] = *(const unsigned*)(arow + half * 16 + 8 + 2 * i);
    }
#pragma unroll
    for (int i = 0; i < 8; ++i) {
      b0.u[i] = *(const unsigned*)(brow0 + half * 16 + 2 * i);
      b1.u[i] = *(const unsigned*)(brow1 + half * 16 + 2 * i);
    }
    acc0 = __builtin_amdgcn_wmma_f32_16x16x32_bf16(
        false, a.v, false, b0.v, (short)0, acc0, false, false);
    acc1 = __builtin_amdgcn_wmma_f32_16x16x32_bf16(
        false, a.v, false, b1.v, (short)0, acc1, false, false);
    __syncthreads();
  }

  // ---- epilogue: BN + leaky ReLU, store bf16 activation [m][OCP] ----
#pragma unroll
  for (int r = 0; r < 8; ++r) {
    int ml = wm * 16 + r + half * 8;   // C layout: vgpr r -> M=r / r+8 per half
    int m  = m0 + ml;
#pragma unroll
    for (int t = 0; t < 2; ++t) {
      int oc = n0 + wn * 32 + t * 16 + l16;
      float cv = t == 0 ? acc0[r] : acc1[r];
      float val = 0.0f;
      if (oc < OC) {
        float s = bn_g[oc] * rsqrtf(bn_v[oc] + 1e-5f);
        val = (cv - bn_m[oc]) * s + bn_b[oc];
        val = val >= 0.0f ? val : 0.1f * val;
      }
      y[(size_t)m * OCP + oc] = f2bf(val);
    }
  }
}

// ---------------------------------------------------------------------------
// Conv2 (1x1) GEMM + per-group softmax over the 9 taps + mask-weighted
// 3x3 unfold aggregation, fully fused. A tile staged with async-to-LDS.
// ---------------------------------------------------------------------------
template <int KD2, int NM, int GROUPS, int CPG, int ICX>
__global__ __launch_bounds__(256)
void conv2_softmax_agg_wmma(const unsigned short* __restrict__ yact,  // [M][KD2] bf16
                            const unsigned short* __restrict__ wt,    // [NM][KD2] bf16
                            const float* __restrict__ x,              // original input
                            float* __restrict__ out8x) {              // [B][ICX][HO][WO]
  constexpr int AS = KD2 + 2;
  constexpr int ZS = NM + 1;
  __shared__ unsigned short As[32 * AS];
  __shared__ float          Zs[32 * ZS];

  const int tid = threadIdx.x;
  const int m0  = blockIdx.x * 32;

  // ---- stage A tile via per-lane async b32 (ASYNCcnt), keeping pad stride ----
  {
    const unsigned abase = lds_addr32(&As[0]);
    const unsigned* ysrc = (const unsigned*)(yact + (size_t)m0 * KD2);
    for (int i = tid; i < 32 * (KD2 / 2); i += 256) {
      int r = i / (KD2 / 2);
      int c = i - r * (KD2 / 2);
      async_b32(abase + (unsigned)(r * AS + 2 * c) * 2u, ysrc + i);
    }
    wait_async0();
  }
  __syncthreads();

  const int lane = tid & 31, wave = tid >> 5;
  const int half = lane >> 4, l16 = lane & 15;
  constexpr int NT = NM / 16;

  for (int t = wave; t < 2 * NT; t += 8) {
    int tm = t & 1;
    int tn = t >> 1;
    v8f acc = {};
    const unsigned short* arow = &As[(tm * 16 + l16) * AS];
    const unsigned short* wrow = &wt[(size_t)(tn * 16 + l16) * KD2];
    for (int ks = 0; ks < KD2 / 32; ++ks) {
      Frag16 a, b;
#pragma unroll
      for (int i = 0; i < 4; ++i) {
        a.u[i]     = *(const unsigned*)(arow + ks * 32 + half * 16 + 2 * i);
        a.u[4 + i] = *(const unsigned*)(arow + ks * 32 + half * 16 + 8 + 2 * i);
      }
#pragma unroll
      for (int i = 0; i < 8; ++i)
        b.u[i] = *(const unsigned*)(wrow + ks * 32 + half * 16 + 2 * i);
      acc = __builtin_amdgcn_wmma_f32_16x16x32_bf16(
          false, a.v, false, b.v, (short)0, acc, false, false);
    }
#pragma unroll
    for (int r = 0; r < 8; ++r)
      Zs[(tm * 16 + r + half * 8) * ZS + tn * 16 + l16] = acc[r];
  }
  __syncthreads();

  // ---- softmax over 9 taps per (row, group) + weighted unfold sum ----
  for (int task = tid; task < 32 * GROUPS; task += 256) {
    int mr = task / GROUPS;
    int g  = task - mr * GROUPS;
    float p[9];
    float mx = -1e30f;
#pragma unroll
    for (int kk = 0; kk < 9; ++kk) {
      p[kk] = Zs[mr * ZS + g * 9 + kk];
      mx = fmaxf(mx, p[kk]);
    }
    float sum = 0.0f;
#pragma unroll
    for (int kk = 0; kk < 9; ++kk) { p[kk] = __expf(p[kk] - mx); sum += p[kk]; }
    float inv = 1.0f / sum;

    int m   = m0 + mr;
    int n   = m / (HO * WO);
    int rem = m - n * (HO * WO);
    int ho  = rem / WO;
    int wo  = rem - ho * WO;
    for (int c1 = 0; c1 < CPG; ++c1) {
      int ch = c1 * GROUPS + g;
      float a = 0.0f;
#pragma unroll
      for (int kk = 0; kk < 9; ++kk) {
        int ih = 2 * ho - 1 + kk / 3;
        int iw = 2 * wo - 1 + kk % 3;
        float xv = ((unsigned)ih < (unsigned)HIMG && (unsigned)iw < (unsigned)WIMG)
                       ? x[((n * ICX + ch) * HIMG + ih) * WIMG + iw] : 0.0f;
        a += (p[kk] * inv) * xv;
      }
      out8x[((n * ICX + ch) * HO + ho) * WO + wo] = a;
    }
  }
}

// ---------------------------------------------------------------------------
// Cost volumes: one block per (b, h) row. g_8x / c_8x rows staged via async
// b128 so each channel row is read once from HBM but reused across all 24
// disparities. Out: out[b][0:8]=gwc corr, out[b][8:32]=concat, [24][80][144].
// ---------------------------------------------------------------------------
__global__ __launch_bounds__(256)
void build_volumes(const float* __restrict__ g8x,   // [4][96][HO][WO]
                   const float* __restrict__ c8x,   // [4][12][HO][WO]
                   float* __restrict__ out) {
  __shared__ float Ls[48 * WO];
  __shared__ float Rs[48 * WO];
  const int h   = blockIdx.x;
  const int b   = blockIdx.y;  // 0..1 (left image b, right image b+2)
  const int tid = threadIdx.x;
  const unsigned lbase = lds_addr32(&Ls[0]);
  const unsigned rbase = lds_addr32(&Rs[0]);

  // two passes of 48 gwc channels (groups 0..3 then 4..7) to bound LDS
  for (int halfc = 0; halfc < 2; ++halfc) {
    for (int i = tid * 4; i < 48 * WO; i += 256 * 4) {  // 4 floats = b128
      int c = i / WO, w = i - c * WO;                   // WO%4==0: stays in row
      int ch = halfc * 48 + c;
      async_b128(lbase + (unsigned)i * 4u,
                 &g8x[(((size_t)b * ICG + ch) * HO + h) * WO + w]);
      async_b128(rbase + (unsigned)i * 4u,
                 &g8x[(((size_t)(b + 2) * ICG + ch) * HO + h) * WO + w]);
    }
    wait_async0();
    __syncthreads();
    for (int i = tid; i < 4 * DISP * WO; i += 256) {
      int gg  = i / (DISP * WO);
      int rem = i - gg * (DISP * WO);
      int d   = rem / WO;
      int w   = rem - d * WO;
      float acc = 0.0f;
      if (w >= d) {
#pragma unroll
        for (int c = 0; c < 12; ++c)
          acc += Ls[(gg * 12 + c) * WO + w] * Rs[(gg * 12 + c) * WO + w - d];
        acc *= (1.0f / 12.0f);
      }
      int g = halfc * 4 + gg;
      out[((((size_t)b * 32 + g) * DISP + d) * HO + h) * WO + w] = acc;
    }
    __syncthreads();
  }

  // concat volume (12 + 12 channels)
  for (int i = tid * 4; i < 12 * WO; i += 256 * 4) {
    int c = i / WO, w = i - c * WO;
    async_b128(lbase + (unsigned)i * 4u,
               &c8x[(((size_t)b * ICC + c) * HO + h) * WO + w]);
    async_b128(rbase + (unsigned)i * 4u,
               &c8x[(((size_t)(b + 2) * ICC + c) * HO + h) * WO + w]);
  }
  wait_async0();
  __syncthreads();
  for (int i = tid; i < 24 * DISP * WO; i += 256) {
    int ch  = i / (DISP * WO);
    int rem = i - ch * (DISP * WO);
    int d   = rem / WO;
    int w   = rem - d * WO;
    float v = 0.0f;
    if (w >= d) v = (ch < 12) ? Ls[ch * WO + w] : Rs[(ch - 12) * WO + w - d];
    out[((((size_t)b * 32 + 8 + ch) * DISP + d) * HO + h) * WO + w] = v;
  }
}

// ---------------------------------------------------------------------------
extern "C" void kernel_launch(void* const* d_in, const int* in_sizes, int n_in,
                              void* d_out, int out_size, void* d_ws, size_t ws_size,
                              hipStream_t stream) {
  (void)in_sizes; (void)n_in; (void)out_size; (void)ws_size;

  const float* gwc_x    = (const float*)d_in[0];
  const float* cat_x    = (const float*)d_in[1];
  const float* gwc_w1   = (const float*)d_in[2];
  const float* gwc_bn_g = (const float*)d_in[3];
  const float* gwc_bn_b = (const float*)d_in[4];
  const float* gwc_bn_m = (const float*)d_in[5];
  const float* gwc_bn_v = (const float*)d_in[6];
  const float* gwc_w2   = (const float*)d_in[7];
  const float* cat_w1   = (const float*)d_in[8];
  const float* cat_bn_g = (const float*)d_in[9];
  const float* cat_bn_b = (const float*)d_in[10];
  const float* cat_bn_m = (const float*)d_in[11];
  const float* cat_bn_v = (const float*)d_in[12];
  const float* cat_w2   = (const float*)d_in[13];
  float* out = (float*)d_out;

  // Workspace carve-up (byte offsets stay 16B-aligned for async b128)
  unsigned short* Wg1 = (unsigned short*)d_ws;              // [192][864]
  unsigned short* Wc1 = Wg1 + 192 * 864;                    // [64][128]
  unsigned short* Wg2 = Wc1 + 64 * 128;                     // [144][192]
  unsigned short* Wc2 = Wg2 + 144 * 192;                    // [112][64]
  unsigned short* Yg  = Wc2 + 112 * 64;                     // [MTOT][192]
  unsigned short* Yc  = Yg + (size_t)MTOT * 192;            // [MTOT][64]
  float* G8X = (float*)(Yc + (size_t)MTOT * 64);            // [4][96][80][144]
  float* C8X = G8X + (size_t)4 * 96 * HO * WO;              // [4][12][80][144]

  // 1) weights -> bf16 (padded where needed)
  convert_pad_bf16<<<(192 * 864 + 255) / 256, 256, 0, stream>>>(gwc_w1, Wg1, 192, 864, 864, 192 * 864);
  convert_pad_bf16<<<(64 * 128 + 255) / 256, 256, 0, stream>>>(cat_w1, Wc1, 24, 108, 128, 64 * 128);
  convert_pad_bf16<<<(144 * 192 + 255) / 256, 256, 0, stream>>>(gwc_w2, Wg2, 144, 192, 192, 144 * 192);
  convert_pad_bf16<<<(112 * 64 + 255) / 256, 256, 0, stream>>>(cat_w2, Wc2, 108, 24, 64, 112 * 64);

  // 2) conv1 + BN + leaky ReLU (implicit GEMM, WMMA bf16, TDM weight tiles)
  conv1_bn_lrelu_wmma<96, 192, 192, 864, 864>
      <<<dim3(MTOT / 64, 3), 256, 0, stream>>>(gwc_x, Wg1, gwc_bn_g, gwc_bn_b,
                                               gwc_bn_m, gwc_bn_v, Yg);
  conv1_bn_lrelu_wmma<12, 24, 64, 108, 128>
      <<<dim3(MTOT / 64, 1), 256, 0, stream>>>(cat_x, Wc1, cat_bn_g, cat_bn_b,
                                               cat_bn_m, cat_bn_v, Yc);

  // 3) conv2 (1x1) + softmax(9) + mask-weighted unfold, fused
  conv2_softmax_agg_wmma<192, 144, 16, 6, 96>
      <<<MTOT / 32, 256, 0, stream>>>(Yg, Wg2, gwc_x, G8X);
  conv2_softmax_agg_wmma<64, 112, 12, 1, 12>
      <<<MTOT / 32, 256, 0, stream>>>(Yc, Wc2, cat_x, C8X);

  // 4) gwc correlation + concat volumes
  build_volumes<<<dim3(HO, 2), 256, 0, stream>>>(G8X, C8X, out);
}
